// MoE_60017872995074
// MI455X (gfx1250) — matrix operand order repaired and checked by reading, not compiled, not run on hardware
//
#include <hip/hip_runtime.h>
#include <hip/hip_bf16.h>

// MoE top-1 SwiGLU, CDNA5 (gfx1250), wave32, bf16 WMMA (v_wmma_f32_16x16x32_bf16)
// Double-buffered LDS weight staging; staging loads use sgpr-base + 32-bit offset
// addressing (global_load), converted to packed bf16 immediately (short liveness),
// ds_store_b64 into the alternate buffer after the WMMA block; 1 barrier / K-step.

typedef __bf16 bf16;
typedef __attribute__((ext_vector_type(16))) __bf16 v16bf;
typedef __attribute__((ext_vector_type(8)))  __bf16 v8bf;
typedef __attribute__((ext_vector_type(4)))  __bf16 v4bf;
typedef __attribute__((ext_vector_type(8)))  float  v8f;

static constexpr int Bb   = 2;
static constexpr int S    = 2048;
static constexpr int D    = 1024;
static constexpr int E    = 4;
static constexpr int H    = 4 * D;      // 4096
static constexpr int NTOK = Bb * S;     // 4096
static constexpr int TILE_M  = 64;      // tokens per block
static constexpr int TILE_N  = 128;     // gate cols per block (gemm1)
static constexpr int TILE_N2 = 128;     // out cols per block (gemm2)
static constexpr int KSTEP   = 32;      // bf16 WMMA K

// ---------------------------------------------------------------- router
__global__ void __launch_bounds__(256)
router_kernel(const float* __restrict__ x, const float* __restrict__ wr,
              int* __restrict__ expert_idx) {
  int wid = threadIdx.x >> 5, lane = threadIdx.x & 31;
  int token = blockIdx.x * 8 + wid;
  if (token >= NTOK) return;
  const float* xr = x + (size_t)token * D;
  float acc[E] = {0.f, 0.f, 0.f, 0.f};
  for (int d = lane; d < D; d += 32) {
    float xv = xr[d];
#pragma unroll
    for (int e = 0; e < E; ++e) acc[e] += xv * wr[d * E + e];
  }
#pragma unroll
  for (int e = 0; e < E; ++e)
#pragma unroll
    for (int off = 16; off > 0; off >>= 1)
      acc[e] += __shfl_xor(acc[e], off, 32);
  if (lane == 0) {
    int best = 0; float bv = acc[0];
#pragma unroll
    for (int e = 1; e < E; ++e) if (acc[e] > bv) { bv = acc[e]; best = e; }
    expert_idx[token] = best;   // softmax is monotone -> argmax(logits)
  }
}

__global__ void zero_counts(int* counts) { if (threadIdx.x < E) counts[threadIdx.x] = 0; }

__global__ void __launch_bounds__(256)
scatter_kernel(const int* __restrict__ expert_idx, int* __restrict__ counts,
               int* __restrict__ list) {
  int t = blockIdx.x * blockDim.x + threadIdx.x;
  if (t >= NTOK) return;
  int e = expert_idx[t];
  int slot = atomicAdd(&counts[e], 1);
  list[e * NTOK + slot] = t;
}

// ------------------------------------------------- GEMM1 + SwiGLU (x @ Wv)
__global__ void __launch_bounds__(256)
gemm1_swiglu(const float* __restrict__ x, const float* __restrict__ wv,
             const int* __restrict__ counts, const int* __restrict__ list,
             bf16* __restrict__ hbuf) {
  const int e = blockIdx.z;
  const int cnt = counts[e];
  const int rowTile = blockIdx.y;
  if (rowTile * TILE_M >= cnt) return;
  const int colBase = blockIdx.x * TILE_N;

  __shared__ bf16 Bs[2][KSTEP][2 * TILE_N];  // double buffer, 2 x 16KB
  __shared__ int  toks[TILE_M];

  const int tid = threadIdx.x;
  if (tid < TILE_M) {
    int s = rowTile * TILE_M + tid;
    toks[tid] = (s < cnt) ? list[e * NTOK + s] : -1;
  }
  __syncthreads();

  const int wid = tid >> 5, lane = tid & 31;
  const int wm = wid >> 1, wn = wid & 1;
  const int half = lane >> 4, ml = lane & 15;
  const int myRow = wm * 16 + ml;               // lanes L and L+16 share A row
  const int tokA = toks[myRow];
  const float* arow = x + (size_t)((tokA < 0) ? 0 : tokA) * D;

  const float* wve = wv + (size_t)e * D * (2 * H);

  // per-thread staging map: 8 float4 per 32x256 tile; 32-bit element offsets
  int pre[8], kks[8], ccs[8];
#pragma unroll
  for (int i = 0; i < 8; ++i) {
    int idx = (tid + i * 256) * 4;              // 0..8188
    kks[i] = idx >> 8; ccs[i] = idx & 255;
    int gcol = (ccs[i] < TILE_N) ? (colBase + ccs[i])
                                 : (H + colBase + (ccs[i] - TILE_N));
    pre[i] = kks[i] * (2 * H) + gcol;           // fits 32-bit
  }

  v8f accG[4] = {}; v8f accV[4] = {};

  // prologue: stage tile 0 into buffer 0
  {
    v4bf pk[8];
#pragma unroll
    for (int i = 0; i < 8; ++i) {
      float4 f = *(const float4*)(wve + (size_t)pre[i]);
      pk[i][0] = (bf16)f.x; pk[i][1] = (bf16)f.y;
      pk[i][2] = (bf16)f.z; pk[i][3] = (bf16)f.w;
    }
#pragma unroll
    for (int i = 0; i < 8; ++i)
      *(v4bf*)&Bs[0][kks[i]][ccs[i]] = pk[i];
  }
  __syncthreads();

  const int NIT = D / KSTEP;                    // 32
  const int krow = ml + half * 16;              // B: lane -> N column ml, 16 K values per lane (krow unused)
  int tileOff = 0;                              // element offset of current K tile
  for (int it = 0; it < NIT; ++it) {
    const int p = it & 1;
    const bool hasNext = (it + 1) < NIT;

    // issue next tile's loads; convert immediately so only packed bf16 stays live
    v4bf pk[8];
    if (hasNext) {
      const int nextOff = tileOff + KSTEP * (2 * H);
#pragma unroll
      for (int i = 0; i < 8; ++i) {
        float4 f = *(const float4*)(wve + (size_t)(nextOff + pre[i]));
        pk[i][0] = (bf16)f.x; pk[i][1] = (bf16)f.y;
        pk[i][2] = (bf16)f.z; pk[i][3] = (bf16)f.w;
      }
    }

    // A fragment 16x32 bf16: lanes 0-15 hold K{0..7,16..23}, lanes 16-31 K{8..15,24..31}
    v16bf a;
    {
      const int kb = it * KSTEP + half * 8;
      float4 f0 = *(const float4*)(arow + kb);
      float4 f1 = *(const float4*)(arow + kb + 4);
      float4 f2 = *(const float4*)(arow + kb + 16);
      float4 f3 = *(const float4*)(arow + kb + 20);
      a[0]=(bf16)f0.x; a[1]=(bf16)f0.y; a[2]=(bf16)f0.z; a[3]=(bf16)f0.w;
      a[4]=(bf16)f1.x; a[5]=(bf16)f1.y; a[6]=(bf16)f1.z; a[7]=(bf16)f1.w;
      a[8]=(bf16)f2.x; a[9]=(bf16)f2.y; a[10]=(bf16)f2.z; a[11]=(bf16)f2.w;
      a[12]=(bf16)f3.x; a[13]=(bf16)f3.y; a[14]=(bf16)f3.z; a[15]=(bf16)f3.w;
    }

#pragma unroll
    for (int t = 0; t < 4; ++t) {
      int c0 = wn * 64 + t * 16;
      v16bf bg;
      for (int s = 0; s < 16; ++s) bg[s] = Bs[p][(s < 8) ? half * 8 + s : 8 + half * 8 + s][c0 + ml];
      accG[t] = __builtin_amdgcn_wmma_f32_16x16x32_bf16(
          false, a, false, bg, (short)0, accG[t], false, false);
      v16bf bv;
      for (int s = 0; s < 16; ++s) bv[s] = Bs[p][(s < 8) ? half * 8 + s : 8 + half * 8 + s][TILE_N + c0 + ml];
      accV[t] = __builtin_amdgcn_wmma_f32_16x16x32_bf16(
          false, a, false, bv, (short)0, accV[t], false, false);
    }

    if (hasNext) {
#pragma unroll
      for (int i = 0; i < 8; ++i)
        *(v4bf*)&Bs[1 - p][kks[i]][ccs[i]] = pk[i];
      tileOff += KSTEP * (2 * H);
    }
    __syncthreads();
  }

  // C layout: VGPR i -> M = half*8 + i, N = ml; SwiGLU then store h (bf16)
#pragma unroll
  for (int t = 0; t < 4; ++t) {
    int col = colBase + wn * 64 + t * 16 + ml;
#pragma unroll
    for (int i = 0; i < 8; ++i) {
      int r = wm * 16 + half * 8 + i;
      int tk = toks[r];
      if (tk >= 0) {
        float g = accG[t][i], vv = accV[t][i];
        float hval = (g / (1.0f + __expf(-g))) * vv;
        hbuf[(size_t)tk * H + col] = (bf16)hval;
      }
    }
  }
}

// ------------------------------------------------- GEMM2 (h @ Wproj) -> out
__global__ void __launch_bounds__(256)
gemm2_kernel(const bf16* __restrict__ hbuf, const float* __restrict__ wp,
             const int* __restrict__ counts, const int* __restrict__ list,
             float* __restrict__ out) {
  const int e = blockIdx.z;
  const int cnt = counts[e];
  const int rowTile = blockIdx.y;
  if (rowTile * TILE_M >= cnt) return;
  const int colBase = blockIdx.x * TILE_N2;

  __shared__ bf16 Bs[2][KSTEP][TILE_N2];   // double buffer, 2 x 8KB
  __shared__ int  toks[TILE_M];

  const int tid = threadIdx.x;
  if (tid < TILE_M) {
    int s = rowTile * TILE_M + tid;
    toks[tid] = (s < cnt) ? list[e * NTOK + s] : -1;
  }
  __syncthreads();

  const int wid = tid >> 5, lane = tid & 31;
  const int wm = wid >> 1, wn = wid & 1;
  const int half = lane >> 4, ml = lane & 15;
  const int myRow = wm * 16 + ml;
  const int tokA = toks[myRow];
  const bf16* hrow = hbuf + (size_t)((tokA < 0) ? 0 : tokA) * H;

  const float* wpe = wp + (size_t)e * H * D;

  // per-thread staging map: 4 float4 per 32x128 tile; 32-bit element offsets
  int pre[4], kks[4], ccs[4];
#pragma unroll
  for (int i = 0; i < 4; ++i) {
    int idx = (tid + i * 256) * 4;              // 0..4092
    kks[i] = idx >> 7; ccs[i] = idx & 127;
    pre[i] = kks[i] * D + colBase + ccs[i];
  }

  v8f acc[4] = {};

  // prologue: stage tile 0 into buffer 0
  {
    v4bf pk[4];
#pragma unroll
    for (int i = 0; i < 4; ++i) {
      float4 f = *(const float4*)(wpe + (size_t)pre[i]);
      pk[i][0] = (bf16)f.x; pk[i][1] = (bf16)f.y;
      pk[i][2] = (bf16)f.z; pk[i][3] = (bf16)f.w;
    }
#pragma unroll
    for (int i = 0; i < 4; ++i)
      *(v4bf*)&Bs[0][kks[i]][ccs[i]] = pk[i];
  }
  __syncthreads();

  const int NIT = H / KSTEP;                    // 128
  const int krow = ml + half * 16;
  int tileOff = 0;
  for (int it = 0; it < NIT; ++it) {
    const int p = it & 1;
    const bool hasNext = (it + 1) < NIT;

    v4bf pk[4];
    if (hasNext) {
      const int nextOff = tileOff + KSTEP * D;
#pragma unroll
      for (int i = 0; i < 4; ++i) {
        float4 f = *(const float4*)(wpe + (size_t)(nextOff + pre[i]));
        pk[i][0] = (bf16)f.x; pk[i][1] = (bf16)f.y;
        pk[i][2] = (bf16)f.z; pk[i][3] = (bf16)f.w;
      }
    }

    v16bf a;
    {
      const int kb = it * KSTEP + half * 8;
      v8bf h0 = *(const v8bf*)(hrow + kb);
      v8bf h1 = *(const v8bf*)(hrow + kb + 16);
#pragma unroll
      for (int j = 0; j < 8; ++j) { a[j] = h0[j]; a[8 + j] = h1[j]; }
    }

#pragma unroll
    for (int t = 0; t < 4; ++t) {
      int c0 = wn * 64 + t * 16;
      v16bf b;
      for (int s = 0; s < 16; ++s) b[s] = Bs[p][(s < 8) ? half * 8 + s : 8 + half * 8 + s][c0 + ml];
      acc[t] = __builtin_amdgcn_wmma_f32_16x16x32_bf16(
          false, a, false, b, (short)0, acc[t], false, false);
    }

    if (hasNext) {
#pragma unroll
      for (int i = 0; i < 4; ++i)
        *(v4bf*)&Bs[1 - p][kks[i]][ccs[i]] = pk[i];
      tileOff += KSTEP * D;
    }
    __syncthreads();
  }

#pragma unroll
  for (int t = 0; t < 4; ++t) {
    int col = colBase + wn * 64 + t * 16 + ml;
#pragma unroll
    for (int i = 0; i < 8; ++i) {
      int r = wm * 16 + half * 8 + i;
      int tk = toks[r];
      if (tk >= 0) out[(size_t)tk * D + col] = acc[t][i];
    }
  }
}

// ---------------------------------------------------------------- launch
extern "C" void kernel_launch(void* const* d_in, const int* in_sizes, int n_in,
                              void* d_out, int out_size, void* d_ws, size_t ws_size,
                              hipStream_t stream) {
  const float* x  = (const float*)d_in[0];
  const float* wr = (const float*)d_in[1];
  const float* wv = (const float*)d_in[2];
  const float* wp = (const float*)d_in[3];
  float* out = (float*)d_out;

  // workspace layout
  int* expert_idx = (int*)d_ws;                         // NTOK ints
  int* counts     = expert_idx + NTOK;                  // E ints (padded)
  int* list       = counts + 64;                        // E*NTOK ints
  bf16* hbuf      = (bf16*)((char*)d_ws + (1 << 17));   // NTOK*H bf16 = 32MB

  router_kernel<<<NTOK / 8, 256, 0, stream>>>(x, wr, expert_idx);
  zero_counts<<<1, 64, 0, stream>>>(counts);
  scatter_kernel<<<NTOK / 256, 256, 0, stream>>>(expert_idx, counts, list);

  gemm1_swiglu<<<dim3(H / TILE_N, NTOK / TILE_M, E), 256, 0, stream>>>(
      x, wv, counts, list, hbuf);
  gemm2_kernel<<<dim3(D / TILE_N2, NTOK / TILE_M, E), 256, 0, stream>>>(
      hbuf, wp, counts, list, out);
}
